// RandLANet_541165879354
// MI455X (gfx1250) — matrix operand order repaired and checked by reading, not compiled, not run on hardware
//
#include <hip/hip_runtime.h>

#define BATCH 2
#define NPTS  8192
#define EPSB  1e-6f

typedef __attribute__((ext_vector_type(16))) _Float16 v16h;
typedef __attribute__((ext_vector_type(8)))  float    v8f;

__device__ __forceinline__ int permN(int i) { return (3413 * i + 1597) & (NPTS - 1); }

__device__ __forceinline__ unsigned ldsOffsetOf(const void* p)
{
    // LDS aperture generic pointers carry the LDS byte offset in the low 32 bits
    return (unsigned)(unsigned long long)p;
}

// ---------------------------------------------------------------------------
// Fused GEMM: Y[b][M][Nc] = act( bn( W @ X[b] + bias ) )
//   W: [M,K] row-major (or [K,M] if TRANSW)   X: [B][K][Nc]   Y: [B][M][Nc]
//   act: 0=none 1=relu 2=leaky0.2 3=leaky0.01
// LDS tiles stored directly in WMMA operand layout so each lane reads its
// whole operand as one aligned v16h (-> ds_load_b128 pairs). Each wave owns a
// 16x32 output slab: two v_wmma_f32_16x16x32_f16 per K-step sharing one A.
// ---------------------------------------------------------------------------
template <int TRANSW>
__global__ __launch_bounds__(128)
void gemm_wmma_kernel(const float* __restrict__ W, const float* __restrict__ X,
                      float* __restrict__ Y,
                      const float* __restrict__ bias,
                      const float* __restrict__ gamma,
                      const float* __restrict__ beta,
                      int M, int K, int Ncols, int act)
{
    // operand-layout tiles: [lane][elem]
    __shared__ __align__(32) _Float16 sA[32][16];
    __shared__ __align__(32) _Float16 sB[4][2][32][16];

    const int tid  = threadIdx.x;
    const int wave = tid >> 5;
    const int lane = tid & 31;
    const int b    = blockIdx.z;
    const int mt   = blockIdx.y * 16;
    const int nt   = (blockIdx.x * 8 + wave * 2) * 16;   // two 16-col tiles

#if __has_builtin(__builtin_amdgcn_s_wait_tensorcnt)
    __builtin_amdgcn_s_wait_tensorcnt(0);
#endif

    const float* Xb = X + (long long)b * K * Ncols;
    float*       Yb = Y + (long long)b * M * Ncols;

    v8f acc0 = {}, acc1 = {};

    for (int k0 = 0; k0 < K; k0 += 32) {
        // ---- stage A (16x32 weight tile) into A-operand layout ----
        // element (r, kk) -> lane = r + 16*(kk>=16), elem = kk&15
        for (int e = tid; e < 512; e += 128) {
            int r  = e >> 5, kk = e & 31;
            int m  = mt + r, k = k0 + kk;
            int mc = min(m, M - 1), kc = min(k, K - 1);
            float w = TRANSW ? W[(long long)kc * M + mc] : W[(long long)mc * K + kc];
            w = (m < M && k < K) ? w : 0.f;
            int lA = r + ((kk >= 16) ? 16 : 0);
            int eA = kk & 15;
            sA[lA][eA] = (_Float16)w;
        }
        // ---- stage B (two 32x16 activation tiles) into B-operand layout ----
        // element (kk, nn) -> lane = nn + 16*(kk>=16), elem = kk&15
        for (int e = lane; e < 1024; e += 32) {
            int t  = e >> 9;
            int kk = (e >> 4) & 31;
            int nn = e & 15;
            int k  = k0 + kk, n = nt + t * 16 + nn;
            int kc = min(k, K - 1), nc = min(n, Ncols - 1);
            float x = Xb[(long long)kc * Ncols + nc];
            x = (k < K && n < Ncols) ? x : 0.f;
            int lB = nn + ((kk >= 16) ? 16 : 0);
            sB[wave][t][lB][kk & 15] = (_Float16)x;
        }
        if (k0 + 32 < K)   // wave-uniform speculative prefetch of next K slice
            __builtin_prefetch(&Xb[(long long)(k0 + 32) * Ncols + min(nt, Ncols - 1)], 0, 1);
        __syncthreads();

        v16h a  = *(const v16h*)sA[lane];
        v16h b0 = *(const v16h*)sB[wave][0][lane];
        v16h b1 = *(const v16h*)sB[wave][1][lane];
        acc0 = __builtin_amdgcn_wmma_f32_16x16x32_f16(false, a, false, b0,
                                                      (short)0, acc0, false, false);
        acc1 = __builtin_amdgcn_wmma_f32_16x16x32_f16(false, a, false, b1,
                                                      (short)0, acc1, false, false);
        __syncthreads();
    }

    // epilogue: VGPR r -> M = r + (lane>=16 ? 8 : 0), col = lane&15
    const int col   = lane & 15;
    const int mbase = mt + ((lane >= 16) ? 8 : 0);
    const float bnscale = rsqrtf(1.f + EPSB);
#pragma unroll
    for (int t = 0; t < 2; ++t) {
        int n = nt + t * 16 + col;
        if (n < Ncols) {
#pragma unroll
            for (int r = 0; r < 8; ++r) {
                int m = mbase + r;
                if (m < M) {
                    float v = t ? acc1[r] : acc0[r];
                    if (bias)  v += bias[m];
                    if (gamma) v = v * (gamma[m] * bnscale) + beta[m];
                    if (act == 1)      v = fmaxf(v, 0.f);
                    else if (act == 2) v = (v > 0.f) ? v : 0.2f  * v;
                    else if (act == 3) v = (v > 0.f) ? v : 0.01f * v;
                    Yb[(long long)m * Ncols + n] = v;
                }
            }
        }
    }
}

// ---------------------------------------------------------------------------
// fc_start + bn + leaky(0.2) + permutation; also emits permuted coords
// ---------------------------------------------------------------------------
__global__ void fc_start_kernel(const float* __restrict__ pos,
                                const float* __restrict__ xin,
                                const float* __restrict__ W,
                                const float* __restrict__ bias,
                                const float* __restrict__ g,
                                const float* __restrict__ be,
                                float* __restrict__ coords,
                                float* __restrict__ h0, int Nn)
{
    int i = blockIdx.x * blockDim.x + threadIdx.x;
    int b = blockIdx.y;
    if (i >= Nn) return;
    int j = permN(i);
    const float* pp = pos + (long long)(b * Nn + j) * 3;
    const float* xx = xin + (long long)(b * Nn + j) * 3;
    float in6[6] = { pp[0], pp[1], pp[2], xx[0], xx[1], xx[2] };
    float* cb = coords + ((long long)b * Nn + i) * 3;
    cb[0] = pp[0]; cb[1] = pp[1]; cb[2] = pp[2];
    const float s = rsqrtf(1.f + EPSB);
#pragma unroll
    for (int c = 0; c < 12; ++c) {
        float v = bias[c];
#pragma unroll
        for (int k = 0; k < 6; ++k) v += W[c * 6 + k] * in6[k];
        v = v * (g[c] * s) + be[c];
        v = (v > 0.f) ? v : 0.2f * v;
        h0[((long long)b * 12 + c) * Nn + i] = v;
    }
}

// ---------------------------------------------------------------------------
// KNN k=16 (squared distances). Support tiles staged global->LDS with the
// gfx1250 async-LDS path (GLOBAL_LOAD_ASYNC_TO_LDS_B32 + s_wait_asynccnt),
// register insertion sort for top-16.
// ---------------------------------------------------------------------------
__global__ __launch_bounds__(256)
void knn16_kernel(const float* __restrict__ coords, int Nl, int Nfull,
                  int* __restrict__ idx, float* __restrict__ dist)
{
    __shared__ __align__(16) float sc[256 * 3];   // AoS tile: [pt][xyz]
    const int i = blockIdx.x * 256 + threadIdx.x;
    const int b = blockIdx.y;
    const float* cb = coords + (long long)b * Nfull * 3;

    float qx = 0.f, qy = 0.f, qz = 0.f;
    if (i < Nl) { qx = cb[i * 3]; qy = cb[i * 3 + 1]; qz = cb[i * 3 + 2]; }

    float bd[16]; int bi[16];
#pragma unroll
    for (int k = 0; k < 16; ++k) { bd[k] = 3.0e38f; bi[k] = 0; }

    const unsigned ldsbase = ldsOffsetOf(sc);

    for (int t0 = 0; t0 < Nl; t0 += 256) {
        int cnt  = min(256, Nl - t0);
        int nflt = cnt * 3;
        __syncthreads();                          // prior tile reads complete
        // async copy: contiguous cnt*3 floats, direct global -> LDS
        for (int e = threadIdx.x; e < nflt; e += 256) {
            unsigned      loff = ldsbase + (unsigned)e * 4u;
            const float*  gp   = cb + (long long)t0 * 3 + e;
            asm volatile("global_load_async_to_lds_b32 %0, %1, off"
                         :: "v"(loff), "v"(gp) : "memory");
        }
        asm volatile("s_wait_asynccnt 0" ::: "memory");
        __syncthreads();                          // writes visible to all waves
        if (i < Nl) {
            for (int t = 0; t < cnt; ++t) {
                float dx = qx - sc[t * 3 + 0];
                float dy = qy - sc[t * 3 + 1];
                float dz = qz - sc[t * 3 + 2];
                float dn = dx * dx + dy * dy + dz * dz;
                if (dn < bd[15]) {
                    bd[15] = dn; bi[15] = t0 + t;
#pragma unroll
                    for (int j = 15; j > 0; --j) {
                        if (bd[j] < bd[j - 1]) {
                            float td = bd[j]; bd[j] = bd[j - 1]; bd[j - 1] = td;
                            int   ti = bi[j]; bi[j] = bi[j - 1]; bi[j - 1] = ti;
                        }
                    }
                }
            }
        }
    }
    if (i < Nl) {
        long long base = ((long long)b * Nl + i) * 16;
#pragma unroll
        for (int k = 0; k < 16; ++k) {
            idx[base + k]  = bi[k];
            dist[base + k] = fmaxf(bd[k], 0.f);
        }
    }
}

// ---------------------------------------------------------------------------
// build f10 = [center(3), neigh(3), center-neigh(3), dist(1)] -> [B][10][Nl*16]
// ---------------------------------------------------------------------------
__global__ void lse_build_kernel(const float* __restrict__ coords,
                                 const int* __restrict__ idx,
                                 const float* __restrict__ dist,
                                 float* __restrict__ f10, int Nl, int Nfull)
{
    long long e = (long long)blockIdx.x * blockDim.x + threadIdx.x;
    int b = blockIdx.y;
    long long NK = (long long)Nl * 16;
    if (e >= NK) return;
    int n = (int)(e >> 4);
    const float* cb = coords + (long long)b * Nfull * 3;
    int j = idx[(long long)b * NK + e];
    float cx = cb[n * 3], cy = cb[n * 3 + 1], cz = cb[n * 3 + 2];
    float nx = cb[j * 3], ny = cb[j * 3 + 1], nz = cb[j * 3 + 2];
    float* fb = f10 + (long long)b * 10 * NK;
    fb[0 * NK + e] = cx; fb[1 * NK + e] = cy; fb[2 * NK + e] = cz;
    fb[3 * NK + e] = nx; fb[4 * NK + e] = ny; fb[5 * NK + e] = nz;
    fb[6 * NK + e] = cx - nx; fb[7 * NK + e] = cy - ny; fb[8 * NK + e] = cz - nz;
    fb[9 * NK + e] = dist[(long long)b * NK + e];
}

// concat enc[hd][NK] with broadcast feats[hd][Nl] -> out[2hd][NK]
__global__ void concat_lse_kernel(const float* __restrict__ enc,
                                  const float* __restrict__ feats,
                                  float* __restrict__ out, int hd, int Nl)
{
    long long NK = (long long)Nl * 16;
    long long tot = 2ll * hd * NK;
    long long i = (long long)blockIdx.x * blockDim.x + threadIdx.x;
    int b = blockIdx.y;
    if (i >= tot) return;
    int c = (int)(i / NK);
    long long e = i - (long long)c * NK;
    float v;
    if (c < hd) v = enc[((long long)b * hd + c) * NK + e];
    else        v = feats[((long long)b * hd + (c - hd)) * Nl + (int)(e >> 4)];
    out[(long long)b * tot + i] = v;
}

// softmax over k=16 of S, then weighted sum of X -> F[C][Nl]
__global__ void attpool_kernel(const float* __restrict__ S,
                               const float* __restrict__ X,
                               float* __restrict__ F, int C, int Nl)
{
    int i = blockIdx.x * blockDim.x + threadIdx.x;
    int b = blockIdx.y;
    if (i >= C * Nl) return;
    int c = i / Nl, n = i - c * Nl;
    long long NK = (long long)Nl * 16;
    const float* Sp = S + ((long long)b * C + c) * NK + (long long)n * 16;
    const float* Xp = X + ((long long)b * C + c) * NK + (long long)n * 16;
    float mx = -3.4e38f;
#pragma unroll
    for (int k = 0; k < 16; ++k) mx = fmaxf(mx, Sp[k]);
    float sum = 0.f, acc = 0.f;
#pragma unroll
    for (int k = 0; k < 16; ++k) {
        float w = __expf(Sp[k] - mx);
        sum += w; acc += w * Xp[k];
    }
    F[((long long)b * C + c) * Nl + n] = acc / sum;
}

__global__ void add_leaky_kernel(const float* __restrict__ A,
                                 const float* __restrict__ B,
                                 float* __restrict__ Y, long long tot)
{
    long long i = (long long)blockIdx.x * blockDim.x + threadIdx.x;
    if (i >= tot) return;
    float v = A[i] + B[i];
    Y[i] = (v > 0.f) ? v : 0.01f * v;
}

__global__ void downsample_kernel(const float* __restrict__ in,
                                  float* __restrict__ out,
                                  int C, int Nin, int Nout)
{
    int i = blockIdx.x * blockDim.x + threadIdx.x;
    int b = blockIdx.y;
    if (i >= C * Nout) return;
    int c = i / Nout, n = i - c * Nout;
    out[((long long)b * C + c) * Nout + n] = in[((long long)b * C + c) * Nin + n];
}

// nearest (k=1) neighbor of each fine point among the coarse set
__global__ void nn1_kernel(const float* __restrict__ coords, int Nsup,
                           int Nqry, int Nfull, int* __restrict__ nnidx)
{
    int q = blockIdx.x * blockDim.x + threadIdx.x;
    int b = blockIdx.y;
    if (q >= Nqry) return;
    const float* cb = coords + (long long)b * Nfull * 3;
    float qx = cb[q * 3], qy = cb[q * 3 + 1], qz = cb[q * 3 + 2];
    float best = 3.4e38f; int bi = 0;
    for (int s = 0; s < Nsup; ++s) {
        float dx = qx - cb[s * 3], dy = qy - cb[s * 3 + 1], dz = qz - cb[s * 3 + 2];
        float d = dx * dx + dy * dy + dz * dz;
        if (d < best) { best = d; bi = s; }
    }
    nnidx[(long long)b * Nqry + q] = bi;
}

// out[C1+C2][Nqry] = concat( h[C1][nnidx], stack[C2][.] )
__global__ void gather_concat_kernel(const float* __restrict__ h,
                                     const int* __restrict__ nnidx,
                                     const float* __restrict__ st,
                                     float* __restrict__ out,
                                     int C1, int C2, int Nsup, int Nqry)
{
    long long tot = (long long)(C1 + C2) * Nqry;
    long long i = (long long)blockIdx.x * blockDim.x + threadIdx.x;
    int b = blockIdx.y;
    if (i >= tot) return;
    int c = (int)(i / Nqry);
    int q = (int)(i - (long long)c * Nqry);
    float v;
    if (c < C1) v = h[((long long)b * C1 + c) * Nsup + nnidx[(long long)b * Nqry + q]];
    else        v = st[((long long)b * C2 + (c - C1)) * Nqry + q];
    out[(long long)b * tot + i] = v;
}

// inverse-permute + transpose to [B*N, 6]
__global__ void output_kernel(const float* __restrict__ hf,
                              float* __restrict__ out, int Nn)
{
    int n = blockIdx.x * blockDim.x + threadIdx.x;
    int b = blockIdx.y;
    if (n >= Nn) return;
    int j = permN(n);
#pragma unroll
    for (int c = 0; c < 6; ++c)
        out[(long long)(b * Nn + j) * 6 + c] = hf[((long long)b * 6 + c) * Nn + n];
}

// ---------------------------------------------------------------------------
// host orchestration
// ---------------------------------------------------------------------------
#define P_ENC(l)      (6 + (l) * 26)
#define P_MID_W       110
#define P_MID_B       111
#define P_DEC(l)      (112 + (l) * 4)
#define P_FC1         128
#define P_FC2         132
#define P_FC3         136

extern "C" void kernel_launch(void* const* d_in, const int* in_sizes, int n_in,
                              void* d_out, int out_size, void* d_ws, size_t ws_size,
                              hipStream_t stream)
{
    (void)in_sizes; (void)n_in; (void)out_size; (void)ws_size;
    const int N = NPTS;

    auto F = [&](int i) { return (const float*)d_in[i]; };

    size_t off = 0;
    auto allocf = [&](size_t nelem) {
        float* p = (float*)((char*)d_ws + off);
        off += ((nelem * sizeof(float) + 255) / 256) * 256;
        return p;
    };
    auto alloci = [&](size_t nelem) {
        int* p = (int*)((char*)d_ws + off);
        off += ((nelem * sizeof(int) + 255) / 256) * 256;
        return p;
    };

    float* coords = allocf((size_t)BATCH * N * 3);
    float* hinbuf = allocf(262144);
    float* h1a    = allocf(262144);
    float* f10    = allocf(2621440);
    float* encb   = allocf(2097152);
    float* catb   = allocf(4194304);
    float* scoreb = allocf(4194304);
    float* poolb  = allocf(524288);
    float* a2b    = allocf(524288);
    float* outb   = allocf(524288);
    float* shortb = allocf(524288);
    float* stk[4];
    stk[0] = allocf(524288); stk[1] = allocf(524288);
    stk[2] = allocf(262144); stk[3] = allocf(131072);
    float* midb   = allocf(32768);
    float* hcatb  = allocf(1048576);
    float* decb0  = allocf(262144);
    float* decb1  = allocf(262144);
    float* fcA    = allocf(1048576);
    float* fcB    = allocf(524288);
    float* fcC    = allocf(98304);
    int*   idxb   = alloci(262144);
    int*   nnidxb = alloci(16384);
    float* distb  = allocf(262144);
    float* h0buf  = allocf(196608);

    auto gemm = [&](const float* W, const float* X, float* Y,
                    const float* bias, const float* g, const float* be,
                    int M, int K, int Nc, int transW, int act) {
        dim3 grid((Nc + 127) / 128, (M + 15) / 16, BATCH);
        if (transW)
            gemm_wmma_kernel<1><<<grid, dim3(128), 0, stream>>>(
                W, X, Y, bias, g, be, M, K, Nc, act);
        else
            gemm_wmma_kernel<0><<<grid, dim3(128), 0, stream>>>(
                W, X, Y, bias, g, be, M, K, Nc, act);
    };
    auto g1d = [&](long long tot) { return dim3((unsigned)((tot + 255) / 256), BATCH); };

    // ---- fc_start + permutation ----
    fc_start_kernel<<<dim3((N + 255) / 256, BATCH), 256, 0, stream>>>(
        F(0), F(1), F(2), F(3), F(4), F(5), coords, h0buf, N);

    // ---- encoder ----
    const int d_in_l[4]  = { 12, 32, 128, 256 };
    const int d_out_l[4] = { 16, 64, 128, 256 };
    const int Nl_l[4]    = { 8192, 2048, 512, 128 };
    const float* hin = h0buf;

    for (int l = 0; l < 4; ++l) {
        const int Nl = Nl_l[l], din = d_in_l[l], dout = d_out_l[l], hd = dout / 2;
        const long long NK = (long long)Nl * 16;
        const int E = P_ENC(l);

        knn16_kernel<<<dim3((Nl + 255) / 256, BATCH), 256, 0, stream>>>(
            coords, Nl, N, idxb, distb);

        gemm(F(E + 0), hin, h1a, F(E + 1), nullptr, nullptr, hd, din, Nl, 0, 2);

        lse_build_kernel<<<g1d(NK), 256, 0, stream>>>(coords, idxb, distb, f10, Nl, N);

        // LSE 1 + att_pool 1
        gemm(F(E + 2), f10, encb, F(E + 3), F(E + 4), F(E + 5), hd, 10, (int)NK, 0, 1);
        concat_lse_kernel<<<g1d(2ll * hd * NK), 256, 0, stream>>>(encb, h1a, catb, hd, Nl);
        gemm(F(E + 6), catb, scoreb, nullptr, nullptr, nullptr, dout, dout, (int)NK, 0, 0);
        attpool_kernel<<<g1d((long long)dout * Nl), 256, 0, stream>>>(scoreb, catb, poolb, dout, Nl);
        gemm(F(E + 7), poolb, h1a, F(E + 8), F(E + 9), F(E + 10), hd, dout, Nl, 0, 1);

        // LSE 2 + att_pool 2
        gemm(F(E + 11), f10, encb, F(E + 12), F(E + 13), F(E + 14), hd, 10, (int)NK, 0, 1);
        concat_lse_kernel<<<g1d(2ll * hd * NK), 256, 0, stream>>>(encb, h1a, catb, hd, Nl);
        gemm(F(E + 15), catb, scoreb, nullptr, nullptr, nullptr, dout, dout, (int)NK, 0, 0);
        attpool_kernel<<<g1d((long long)dout * Nl), 256, 0, stream>>>(scoreb, catb, poolb, dout, Nl);
        gemm(F(E + 16), poolb, a2b, F(E + 17), F(E + 18), F(E + 19), dout, dout, Nl, 0, 1);

        // out + shortcut + leaky(0.01)
        gemm(F(E + 20), a2b, outb, F(E + 21), nullptr, nullptr, 2 * dout, dout, Nl, 0, 0);
        gemm(F(E + 22), hin, shortb, F(E + 23), F(E + 24), F(E + 25), 2 * dout, din, Nl, 0, 0);
        long long tot = (long long)BATCH * 2 * dout * Nl;
        add_leaky_kernel<<<dim3((unsigned)((tot + 255) / 256)), 256, 0, stream>>>(
            outb, shortb, stk[l], tot);

        downsample_kernel<<<g1d((long long)2 * dout * (Nl / 4)), 256, 0, stream>>>(
            stk[l], hinbuf, 2 * dout, Nl, Nl / 4);
        hin = hinbuf;
    }

    // ---- mid ----
    gemm(F(P_MID_W), hinbuf, midb, F(P_MID_B), nullptr, nullptr, 512, 512, 32, 0, 1);

    // ---- decoder ----
    const int C1s[4]  = { 512, 256, 128, 32 };
    const int Mos[4]  = { 256, 128, 32, 12 };
    const int Nsup[4] = { 32, 128, 512, 2048 };
    const int Nqry[4] = { 128, 512, 2048, 8192 };
    float* stks[4]    = { stk[3], stk[2], stk[1], stk[0] };
    float* decouts[4] = { decb0, decb1, decb0, decb1 };
    const float* hdec = midb;

    for (int l = 0; l < 4; ++l) {
        nn1_kernel<<<dim3((Nqry[l] + 255) / 256, BATCH), 256, 0, stream>>>(
            coords, Nsup[l], Nqry[l], N, nnidxb);
        int Kc = 2 * C1s[l];
        gather_concat_kernel<<<g1d((long long)Kc * Nqry[l]), 256, 0, stream>>>(
            hdec, nnidxb, stks[l], hcatb, C1s[l], C1s[l], Nsup[l], Nqry[l]);
        const int D = P_DEC(l);
        gemm(F(D + 0), hcatb, decouts[l], F(D + 1), F(D + 2), F(D + 3),
             Mos[l], Kc, Nqry[l], 1 /*transW*/, 1);
        hdec = decouts[l];
    }

    // ---- head ----
    gemm(F(P_FC1 + 0), hdec, fcA, F(P_FC1 + 1), F(P_FC1 + 2), F(P_FC1 + 3), 64, 12, N, 0, 1);
    gemm(F(P_FC2 + 0), fcA, fcB, F(P_FC2 + 1), F(P_FC2 + 2), F(P_FC2 + 3), 32, 64, N, 0, 1);
    gemm(F(P_FC3 + 0), fcB, fcC, F(P_FC3 + 1), nullptr, nullptr, 6, 32, N, 0, 0);

    output_kernel<<<dim3((N + 255) / 256, BATCH), 256, 0, stream>>>(fcC, (float*)d_out, N);
}
